// enc_mtan_rnn_13297218749061
// MI455X (gfx1250) — compile-verified, matrix operand order checked by reading, not run on hardware
//
#include <hip/hip_runtime.h>
#include <hip/hip_bf16.h>

// ---------------------------------------------------------------------------
// Types
// ---------------------------------------------------------------------------
typedef __attribute__((ext_vector_type(8)))  __bf16 bf16x8;
typedef __attribute__((ext_vector_type(16))) __bf16 bf16x16;
typedef __attribute__((ext_vector_type(8)))  float  f32x8;
typedef __attribute__((ext_vector_type(4)))  int    i32x4;
// Address-space-qualified 128-bit vector types for the async DMA builtin:
// param 0 is AS1 (global) v4i*, param 1 is AS3 (LDS) v4i*.
typedef __attribute__((address_space(1))) i32x4 g_i32x4;
typedef __attribute__((address_space(3))) i32x4 l_i32x4;

// Async global->LDS DMA (gfx1250): guarded so missing builtins fall back to
// plain vector loads. Lowers to global_load_async_to_lds_b128 / s_wait_asynccnt.
#if defined(__has_builtin)
#if __has_builtin(__builtin_amdgcn_global_load_async_to_lds_b128) && \
    __has_builtin(__builtin_amdgcn_s_wait_asynccnt)
#define HAVE_ASYNC_LDS 1
#endif
#endif

// ---------------------------------------------------------------------------
// bf16 <-> f32 helpers (round-to-nearest-even)
// ---------------------------------------------------------------------------
__device__ __forceinline__ __bf16 f2bf(float f) {
  union { float f; unsigned u; } in; in.f = f;
  unsigned u = in.u + 0x7FFFu + ((in.u >> 16) & 1u);
  union { unsigned short s; __bf16 b; } out; out.s = (unsigned short)(u >> 16);
  return out.b;
}
__device__ __forceinline__ float bf2f(__bf16 b) {
  union { unsigned short s; __bf16 b; } in; in.b = b;
  union { unsigned u; float f; } out; out.u = ((unsigned)in.s) << 16;
  return out.f;
}
__device__ __forceinline__ float sigmoidf_(float x) {
  return 1.0f / (1.0f + __expf(-x));
}

// ---------------------------------------------------------------------------
// WMMA tile loader.
// CDNA5 16-bit A-matrix 16x32 layout (B mirrors it with lane = column):
//   lane holds row (lane&15); element i holds K = 16*(i/8) + 8*(lane>>4) + i%8
// => for row-major A (lda = K-dim) a lane needs two contiguous 16B loads at
//    k0 + 8*half and k0 + 16 + 8*half. Same for B^T row-major (weights (N,K)).
// ---------------------------------------------------------------------------
__device__ __forceinline__ bf16x16 load_ab_tile(const __bf16* base, int ld,
                                                int rc0, int k0, int lane) {
  const int rc  = rc0 + (lane & 15);
  const int hlf = (lane >> 4) & 1;
  const __bf16* p = base + (size_t)rc * (size_t)ld + k0 + 8 * hlf;
  bf16x8 lo = *(const bf16x8*)p;
  bf16x8 hi = *(const bf16x8*)(p + 16);
  bf16x16 r;
#pragma unroll
  for (int i = 0; i < 8; ++i) { r[i] = lo[i]; r[i + 8] = hi[i]; }
  return r;
}

#define WMMA_BF16(a, b, c) \
  __builtin_amdgcn_wmma_f32_16x16x32_bf16(false, (a), false, (b), (short)0, (c), false, false)

// ---------------------------------------------------------------------------
// Problem constants
// ---------------------------------------------------------------------------
#define BB   32
#define LL   256
#define LQ   128
#define DD   41
#define CC   82
#define CPAD 96
#define ET   128
#define NH   128

// ---------------------------------------------------------------------------
// Kernel: convert / pad weights to bf16
// ---------------------------------------------------------------------------
__global__ void prep_weights_kernel(
    const float* __restrict__ Wq,   const float* __restrict__ Wk,
    const float* __restrict__ Wo,   const float* __restrict__ wihf,
    const float* __restrict__ whhf, const float* __restrict__ wihb,
    const float* __restrict__ whhb, const float* __restrict__ W1,
    const float* __restrict__ b1,   const float* __restrict__ W2,
    __bf16* wq_b, __bf16* wk_b, __bf16* wo_b, __bf16* wihf_b, __bf16* whhf_b,
    __bf16* wihb_b, __bf16* whhb_b, __bf16* w1p, __bf16* w2p, float* b1p) {
  int i = blockIdx.x * blockDim.x + threadIdx.x;
  if (i < 16384) { wq_b[i] = f2bf(Wq[i]); return; }           i -= 16384;
  if (i < 16384) { wk_b[i] = f2bf(Wk[i]); return; }           i -= 16384;
  if (i < 128 * CPAD) {                                        // Wo pad 82->96
    int n = i / CPAD, k = i % CPAD;
    wo_b[i] = f2bf(k < CC ? Wo[n * CC + k] : 0.0f); return;
  }                                                            i -= 128 * CPAD;
  if (i < 49152) { wihf_b[i] = f2bf(wihf[i]); return; }        i -= 49152;
  if (i < 49152) { whhf_b[i] = f2bf(whhf[i]); return; }        i -= 49152;
  if (i < 49152) { wihb_b[i] = f2bf(wihb[i]); return; }        i -= 49152;
  if (i < 49152) { whhb_b[i] = f2bf(whhb[i]); return; }        i -= 49152;
  if (i < 64 * 256) {                                          // W1 pad 50->64
    int n = i / 256;
    w1p[i] = f2bf(n < 50 ? W1[i] : 0.0f); return;
  }                                                            i -= 64 * 256;
  if (i < 32 * 64) {                                           // W2 pad 50->64
    int n = i / 64, k = i % 64;
    w2p[i] = f2bf(k < 50 ? W2[n * 50 + k] : 0.0f); return;
  }                                                            i -= 32 * 64;
  if (i < 64) { b1p[i] = (i < 50) ? b1[i] : 0.0f; return; }
}
#define PREPW_TOTAL (16384 + 16384 + 128 * CPAD + 4 * 49152 + 64 * 256 + 32 * 64 + 64)

// ---------------------------------------------------------------------------
// Kernel: build transposed, padded X^T and M^T per batch (bf16, [b][c][l])
// x = [vals*msk, msk]; mask channel c is x[..., 41 + c%41]
// ---------------------------------------------------------------------------
__global__ void prep_xt_kernel(const float* __restrict__ x,
                               __bf16* __restrict__ Xt, __bf16* __restrict__ Mt) {
  int i = blockIdx.x * blockDim.x + threadIdx.x;
  if (i >= BB * CPAD * LL) return;
  int l = i % LL;
  int c = (i / LL) % CPAD;
  int b = i / (CPAD * LL);
  float xv = 0.0f, mv = 0.0f;
  if (c < CC) {
    size_t base = ((size_t)b * LL + l) * CC;
    xv = x[base + c];
    mv = x[base + DD + (c % DD)];
  }
  Xt[i] = f2bf(xv);
  Mt[i] = f2bf(mv);
}

// ---------------------------------------------------------------------------
// Kernel: sinusoidal time embeddings for keys (B*L) and queries (B*LQ), bf16
// ang = 48*pos*exp(-j*ln(10)/64), emb interleaved [sin, cos]
// ---------------------------------------------------------------------------
__global__ void embed_kernel(const float* __restrict__ ts, const float* __restrict__ qr,
                             __bf16* __restrict__ kemb, __bf16* __restrict__ qemb) {
  int tid = blockIdx.x * blockDim.x + threadIdx.x;
  const int nk = BB * LL, nq = BB * LQ;
  if (tid >= (nk + nq) * 64) return;
  int j = tid & 63;
  int row = tid >> 6;
  float pos;
  __bf16* out;
  if (row < nk) { pos = ts[row]; out = kemb + (size_t)row * ET; }
  else          { int r2 = row - nk; pos = qr[r2]; out = qemb + (size_t)r2 * ET; }
  float dv  = __expf(-(float)j * 0.03597892416f);  // ln(10)/64
  float ang = 48.0f * pos * dv;
  float s, c;
  __sincosf(ang, &s, &c);
  out[2 * j]     = f2bf(s);
  out[2 * j + 1] = f2bf(c);
}

// ---------------------------------------------------------------------------
// Generic bf16 WMMA GEMM: C[M,N] = act( (A[M,K] @ Bt[N,K]^T + bias) * alpha )
// flags bit0: bf16 output, bit1: relu. M%16==0, N%16==0, K%32==0.
// ---------------------------------------------------------------------------
__global__ void gemm_kernel(const __bf16* __restrict__ A, const __bf16* __restrict__ Bt,
                            const float* __restrict__ bias, void* __restrict__ Cout,
                            int M, int N, int K, int flags, float alpha) {
  const int lane   = threadIdx.x & 31;
  const int wave   = (blockIdx.x * blockDim.x + threadIdx.x) >> 5;
  const int nwaves = (gridDim.x * blockDim.x) >> 5;
  const int nt_n   = N >> 4;
  const int ntiles = (M >> 4) * nt_n;
  const int hlf = lane >> 4, ln = lane & 15;
  for (int t = wave; t < ntiles; t += nwaves) {
    const int mt = t / nt_n, nt = t % nt_n;
    f32x8 acc = {};
    for (int k0 = 0; k0 < K; k0 += 32) {
      bf16x16 a = load_ab_tile(A,  K, mt * 16, k0, lane);
      bf16x16 b = load_ab_tile(Bt, K, nt * 16, k0, lane);
      acc = WMMA_BF16(a, b, acc);
    }
    const int col = nt * 16 + ln;
    const float bv = bias ? bias[col] : 0.0f;
#pragma unroll
    for (int r = 0; r < 8; ++r) {
      const int row = mt * 16 + r + 8 * hlf;
      float v = (acc[r] + bv) * alpha;
      if (flags & 2) v = fmaxf(v, 0.0f);
      if (flags & 1) ((__bf16*)Cout)[(size_t)row * N + col] = f2bf(v);
      else           ((float*)Cout)[(size_t)row * N + col] = v;
    }
  }
}

// ---------------------------------------------------------------------------
// Kernel: per-batch scores S = Q@K^T (scale pre-folded into Q), rowmax, exp
// -> E bf16 [b][q][l]. One block per batch, 8 waves = 8 M-strips of 16 rows.
// Single pass: all 16 N-tile accumulators live in registers (16 x v8f =
// 128 VGPRs/lane), A-tile hoisted out of the N loop; rowmax via 16-lane
// xor shuffles directly on the C-layout registers.
// ---------------------------------------------------------------------------
__global__ void scores_kernel(const __bf16* __restrict__ Q, const __bf16* __restrict__ Kp,
                              __bf16* __restrict__ E) {
  const int b    = blockIdx.x;
  const int lane = threadIdx.x & 31;
  const int wave = threadIdx.x >> 5;
  const int hlf = lane >> 4, ln = lane & 15;
  const __bf16* Qb = Q  + (size_t)b * LQ * ET;
  const __bf16* Kb = Kp + (size_t)b * LL * ET;
  __bf16*       Eb = E  + (size_t)b * LQ * LL;
  const int m0 = wave * 16;

  const f32x8 vzero = {};
  f32x8 acc[LL / 16];
#pragma unroll
  for (int nt = 0; nt < LL / 16; ++nt) acc[nt] = vzero;

#pragma unroll
  for (int k0 = 0; k0 < ET; k0 += 32) {
    bf16x16 a = load_ab_tile(Qb, ET, m0, k0, lane);
#pragma unroll
    for (int nt = 0; nt < LL / 16; ++nt) {
      bf16x16 bt = load_ab_tile(Kb, ET, nt * 16, k0, lane);
      acc[nt] = WMMA_BF16(a, bt, acc[nt]);
    }
  }

  float rmax[8];
#pragma unroll
  for (int r = 0; r < 8; ++r) rmax[r] = -3.0e38f;
#pragma unroll
  for (int nt = 0; nt < LL / 16; ++nt)
#pragma unroll
    for (int r = 0; r < 8; ++r) rmax[r] = fmaxf(rmax[r], acc[nt][r]);
#pragma unroll
  for (int m = 1; m < 16; m <<= 1)
#pragma unroll
    for (int r = 0; r < 8; ++r)
      rmax[r] = fmaxf(rmax[r], __shfl_xor(rmax[r], m, 32));

#pragma unroll
  for (int nt = 0; nt < LL / 16; ++nt)
#pragma unroll
    for (int r = 0; r < 8; ++r) {
      int row = m0 + r + 8 * hlf;
      Eb[(size_t)row * LL + nt * 16 + ln] = f2bf(__expf(acc[nt][r] - rmax[r]));
    }
}

// ---------------------------------------------------------------------------
// Kernel: att = (E@X) / (E@M) per batch -> attb bf16 [b][q][CPAD]
// A-tile hoisted out of the N loop; 6 numerator + 6 denominator accumulators
// live in registers.
// ---------------------------------------------------------------------------
__global__ void att_kernel(const __bf16* __restrict__ E, const __bf16* __restrict__ Xt,
                           const __bf16* __restrict__ Mt, __bf16* __restrict__ attb) {
  const int b    = blockIdx.x;
  const int lane = threadIdx.x & 31;
  const int wave = threadIdx.x >> 5;
  const int hlf = lane >> 4, ln = lane & 15;
  const __bf16* Eb = E    + (size_t)b * LQ * LL;
  const __bf16* Xb = Xt   + (size_t)b * CPAD * LL;
  const __bf16* Mb = Mt   + (size_t)b * CPAD * LL;
  __bf16*       Ab = attb + (size_t)b * LQ * CPAD;
  const int m0 = wave * 16;

  const f32x8 vzero = {};
  f32x8 an[CPAD / 16], ad[CPAD / 16];
#pragma unroll
  for (int nt = 0; nt < CPAD / 16; ++nt) { an[nt] = vzero; ad[nt] = vzero; }

#pragma unroll
  for (int k0 = 0; k0 < LL; k0 += 32) {
    bf16x16 a = load_ab_tile(Eb, LL, m0, k0, lane);
#pragma unroll
    for (int nt = 0; nt < CPAD / 16; ++nt) {
      bf16x16 bx = load_ab_tile(Xb, LL, nt * 16, k0, lane);
      bf16x16 bm = load_ab_tile(Mb, LL, nt * 16, k0, lane);
      an[nt] = WMMA_BF16(a, bx, an[nt]);
      ad[nt] = WMMA_BF16(a, bm, ad[nt]);
    }
  }

#pragma unroll
  for (int nt = 0; nt < CPAD / 16; ++nt)
#pragma unroll
    for (int r = 0; r < 8; ++r) {
      int row = m0 + r + 8 * hlf;
      float v = an[nt][r] / (ad[nt][r] + 1e-12f);
      Ab[(size_t)row * CPAD + nt * 16 + ln] = f2bf(v);
    }
}

// ---------------------------------------------------------------------------
// Kernel: bidirectional GRU. One block per direction (gridDim.x == 2).
// Whh (384x128 bf16 = 96KB) is DMA'd into LDS once via the gfx1250 async
// global->LDS path (ASYNCcnt) when available, so every step's B-tiles come
// from ds_load_b128 instead of round-tripping L2.
// Per step: G = Hbf(32x128) @ Whh^T(384x128) via WMMA, gates computed in the
// WMMA C-register layout. h state lives in LDS (f32 + bf16 mirror).
// Two barriers per step order the LDS RAW/WAR hazards. gx rows for the next
// step are prefetched (global_prefetch_b8) during the gate phase.
// LDS plan (dynamic): Hf 16KB | Hb 8KB | Wsh 96KB  => 120KB total
// (CDNA5 WGP has 320KB LDS).
// ---------------------------------------------------------------------------
#define GRU_LDS_BYTES (BB * NH * 4 + BB * NH * 2 + 384 * NH * 2)

__global__ void gru_kernel(const __bf16* __restrict__ gxf, const __bf16* __restrict__ gxb,
                           const __bf16* __restrict__ whhf, const __bf16* __restrict__ whhb,
                           const float* __restrict__ bhhf, const float* __restrict__ bhhb,
                           __bf16* __restrict__ hcat) {
  extern __shared__ __align__(16) char smem[];
  float*  Hf  = (float*)smem;                              // 32*128 f32
  __bf16* Hb  = (__bf16*)(smem + BB * NH * 4);             // 32*128 bf16
  __bf16* Wsh = (__bf16*)(smem + BB * NH * 4 + BB * NH * 2); // 384*128 bf16

  const int dir = blockIdx.x;
  const __bf16* gx  = dir ? gxb  : gxf;
  const __bf16* whh = dir ? whhb : whhf;
  const float*  bhh = dir ? bhhb : bhhf;

  const int lane = threadIdx.x & 31;
  const int wave = threadIdx.x >> 5;
  const int hlf = lane >> 4, ln = lane & 15;

  // Preload Whh into LDS (16B chunks). Async DMA path overlaps with the
  // h-state zeroing below; completion enforced by s_wait_asynccnt + barrier.
#if defined(HAVE_ASYNC_LDS)
  for (int i = threadIdx.x * 8; i < 384 * NH; i += blockDim.x * 8)
    __builtin_amdgcn_global_load_async_to_lds_b128(
        (g_i32x4*)(whh + i), (l_i32x4*)(Wsh + i), 0, 0);
#else
  for (int i = threadIdx.x * 8; i < 384 * NH; i += blockDim.x * 8)
    *(bf16x8*)(Wsh + i) = *(const bf16x8*)(whh + i);
#endif
  for (int i = threadIdx.x; i < BB * NH; i += blockDim.x) {
    Hf[i] = 0.0f;
    Hb[i] = f2bf(0.0f);
  }
#if defined(HAVE_ASYNC_LDS)
  __builtin_amdgcn_s_wait_asynccnt(0);
#endif
  __syncthreads();

  for (int s = 0; s < LQ; ++s) {
    const int t = dir ? (LQ - 1 - s) : s;
    f32x8 aR[2], aZ[2], aN[2];
#pragma unroll
    for (int gg = 0; gg < 2; ++gg) {
      const int g  = wave * 2 + gg;        // 16 groups: 2 M-tiles x 8 N-tiles
      const int m0 = (g >> 3) * 16;
      const int n0 = (g & 7) * 16;
      f32x8 r = {}, z = {}, n = {};
#pragma unroll
      for (int k0 = 0; k0 < NH; k0 += 32) {
        bf16x16 a  = load_ab_tile(Hb,  NH, m0, k0, lane);
        bf16x16 br = load_ab_tile(Wsh, NH, n0,        k0, lane);
        bf16x16 bz = load_ab_tile(Wsh, NH, NH + n0,   k0, lane);
        bf16x16 bn = load_ab_tile(Wsh, NH, 2*NH + n0, k0, lane);
        r = WMMA_BF16(a, br, r);
        z = WMMA_BF16(a, bz, z);
        n = WMMA_BF16(a, bn, n);
      }
      aR[gg] = r; aZ[gg] = z; aN[gg] = n;
    }
    __syncthreads();   // all waves done reading Hb before anyone rewrites it
#pragma unroll
    for (int gg = 0; gg < 2; ++gg) {
      const int g  = wave * 2 + gg;
      const int m0 = (g >> 3) * 16;
      const int n0 = (g & 7) * 16;
      const int nc = n0 + ln;
      const float brr = bhh[nc], bzz = bhh[NH + nc], bnn = bhh[2 * NH + nc];
#pragma unroll
      for (int r = 0; r < 8; ++r) {
        const int bidx = m0 + r + 8 * hlf;           // batch index 0..31
        const size_t row = (size_t)bidx * LQ + t;    // row in g_in
        const float xr = bf2f(gx[row * 384 + nc]);
        const float xz = bf2f(gx[row * 384 + NH + nc]);
        const float xn = bf2f(gx[row * 384 + 2 * NH + nc]);
        const float hold = Hf[bidx * NH + nc];
        const float rr = sigmoidf_(xr + aR[gg][r] + brr);
        const float zz = sigmoidf_(xz + aZ[gg][r] + bzz);
        const float nn = tanhf(xn + rr * (aN[gg][r] + bnn));
        const float h  = (1.0f - zz) * nn + zz * hold;
        Hf[bidx * NH + nc] = h;
        const __bf16 hb = f2bf(h);
        Hb[bidx * NH + nc] = hb;
        hcat[row * 256 + dir * NH + nc] = hb;
        // hide L2 latency for the next step's gate inputs
        if (s + 1 < LQ) {
          const size_t row2 = (size_t)bidx * LQ + (dir ? (t - 1) : (t + 1));
          __builtin_prefetch((const void*)(gx + row2 * 384 + nc), 0, 1);
        }
      }
    }
    __syncthreads();   // writes visible before next step's reads
  }
}

// ---------------------------------------------------------------------------
// Host launcher
// ---------------------------------------------------------------------------
static inline int gemm_grid(int M, int N) {
  long tiles  = (long)(M / 16) * (N / 16);
  long blocks = (tiles + 7) / 8;       // 8 waves per 256-thread block
  if (blocks > 1024) blocks = 1024;
  if (blocks < 1) blocks = 1;
  return (int)blocks;
}

extern "C" void kernel_launch(void* const* d_in, const int* in_sizes, int n_in,
                              void* d_out, int out_size, void* d_ws, size_t ws_size,
                              hipStream_t stream) {
  (void)in_sizes; (void)n_in; (void)out_size;

  const float* x    = (const float*)d_in[0];
  const float* ts   = (const float*)d_in[1];
  const float* qr   = (const float*)d_in[2];
  const float* Wq   = (const float*)d_in[3];
  const float* bq   = (const float*)d_in[4];
  const float* Wk   = (const float*)d_in[5];
  const float* bk   = (const float*)d_in[6];
  const float* Wo   = (const float*)d_in[7];
  const float* bo   = (const float*)d_in[8];
  const float* wihf = (const float*)d_in[9];
  const float* whhf = (const float*)d_in[10];
  const float* bihf = (const float*)d_in[11];
  const float* bhhf = (const float*)d_in[12];
  const float* wihb = (const float*)d_in[13];
  const float* whhb = (const float*)d_in[14];
  const float* bihb = (const float*)d_in[15];
  const float* bhhb = (const float*)d_in[16];
  const float* W1   = (const float*)d_in[17];
  const float* b1   = (const float*)d_in[18];
  const float* W2   = (const float*)d_in[19];
  const float* b2   = (const float*)d_in[20];
  float* out = (float*)d_out;

  char* w = (char*)d_ws;
  size_t off = 0;
  auto alloc = [&](size_t bytes) -> void* {
    void* p = w + off;
    off = (off + bytes + 255) & ~(size_t)255;
    return p;
  };
  __bf16* wq_b   = (__bf16*)alloc(128 * 128 * 2);
  __bf16* wk_b   = (__bf16*)alloc(128 * 128 * 2);
  __bf16* wo_b   = (__bf16*)alloc(128 * CPAD * 2);
  __bf16* wihf_b = (__bf16*)alloc(384 * 128 * 2);
  __bf16* whhf_b = (__bf16*)alloc(384 * 128 * 2);
  __bf16* wihb_b = (__bf16*)alloc(384 * 128 * 2);
  __bf16* whhb_b = (__bf16*)alloc(384 * 128 * 2);
  __bf16* w1p    = (__bf16*)alloc(64 * 256 * 2);
  __bf16* w2p    = (__bf16*)alloc(32 * 64 * 2);
  float*  b1p    = (float*) alloc(64 * 4);
  __bf16* kemb   = (__bf16*)alloc((size_t)BB * LL * ET * 2);
  __bf16* qemb   = (__bf16*)alloc((size_t)BB * LQ * ET * 2);
  __bf16* kproj  = (__bf16*)alloc((size_t)BB * LL * ET * 2);
  __bf16* qproj  = (__bf16*)alloc((size_t)BB * LQ * ET * 2);
  __bf16* Ebuf   = (__bf16*)alloc((size_t)BB * LQ * LL * 2);
  __bf16* Xt     = (__bf16*)alloc((size_t)BB * CPAD * LL * 2);
  __bf16* Mt     = (__bf16*)alloc((size_t)BB * CPAD * LL * 2);
  __bf16* attb   = (__bf16*)alloc((size_t)BB * LQ * CPAD * 2);
  __bf16* outb   = (__bf16*)alloc((size_t)BB * LQ * NH * 2);
  __bf16* gxfb   = (__bf16*)alloc((size_t)BB * LQ * 384 * 2);
  __bf16* gxbb   = (__bf16*)alloc((size_t)BB * LQ * 384 * 2);
  __bf16* hcat   = (__bf16*)alloc((size_t)BB * LQ * 256 * 2);
  __bf16* a2     = (__bf16*)alloc((size_t)BB * LQ * 64 * 2);
  if (off > ws_size) return;  // workspace too small; bail safely

  const int M4 = BB * LQ;   // 4096 query rows
  const int M8 = BB * LL;   // 8192 key rows

  // 1) weight conversion / padding
  prep_weights_kernel<<<(PREPW_TOTAL + 255) / 256, 256, 0, stream>>>(
      Wq, Wk, Wo, wihf, whhf, wihb, whhb, W1, b1, W2,
      wq_b, wk_b, wo_b, wihf_b, whhf_b, wihb_b, whhb_b, w1p, w2p, b1p);
  // 2) X^T / M^T
  prep_xt_kernel<<<(BB * CPAD * LL + 255) / 256, 256, 0, stream>>>(x, Xt, Mt);
  // 3) time embeddings
  embed_kernel<<<((M8 + M4) * 64 + 255) / 256, 256, 0, stream>>>(ts, qr, kemb, qemb);
  // 4) K projection: kemb @ Wk^T + bk
  gemm_kernel<<<gemm_grid(M8, ET), 256, 0, stream>>>(kemb, wk_b, bk, kproj,
      M8, ET, ET, 1, 1.0f);
  // 5) Q projection (1/sqrt(128) folded in)
  gemm_kernel<<<gemm_grid(M4, ET), 256, 0, stream>>>(qemb, wq_b, bq, qproj,
      M4, ET, ET, 1, 0.08838834764831845f);
  // 6) scores -> E = exp(S - rowmax)
  scores_kernel<<<BB, 256, 0, stream>>>(qproj, kproj, Ebuf);
  // 7) att = (E@X)/(E@M)
  att_kernel<<<BB, 256, 0, stream>>>(Ebuf, Xt, Mt, attb);
  // 8) out = att @ Wo^T + bo
  gemm_kernel<<<gemm_grid(M4, NH), 256, 0, stream>>>(attb, wo_b, bo, outb,
      M4, NH, CPAD, 1, 1.0f);
  // 9) GRU input projections (both directions)
  gemm_kernel<<<gemm_grid(M4, 384), 256, 0, stream>>>(outb, wihf_b, bihf, gxfb,
      M4, 384, NH, 1, 1.0f);
  gemm_kernel<<<gemm_grid(M4, 384), 256, 0, stream>>>(outb, wihb_b, bihb, gxbb,
      M4, 384, NH, 1, 1.0f);
  // 10) bidirectional GRU (2 blocks: fwd, bwd); 120KB dynamic LDS per block
  gru_kernel<<<2, 256, GRU_LDS_BYTES, stream>>>(gxfb, gxbb, whhf_b, whhb_b,
      bhhf, bhhb, hcat);
  // 11) MLP layer 1 (relu, padded 50->64)
  gemm_kernel<<<gemm_grid(M4, 64), 256, 0, stream>>>(hcat, w1p, b1p, a2,
      M4, 64, 256, 1 | 2, 1.0f);
  // 12) MLP layer 2 -> f32 output (B,1,LQ,32)
  gemm_kernel<<<gemm_grid(M4, 32), 256, 0, stream>>>(a2, w2p, b2, (void*)out,
      M4, 32, 64, 0, 1.0f);
}
